// GNN_75685913690122
// MI455X (gfx1250) — matrix operand order repaired and checked by reading, not compiled, not run on hardware
//
#include <hip/hip_runtime.h>

typedef __attribute__((ext_vector_type(16))) _Float16 v16h;
typedef __attribute__((ext_vector_type(8)))  float    v8f;

// ---------------------------------------------------------------- degree/norm
__global__ void k_init_deg(float* deg, int n) {
  int i = blockIdx.x * blockDim.x + threadIdx.x;
  if (i < n) deg[i] = 1.0f;                      // self-loop contribution
}
__global__ void k_count_deg(const int* __restrict__ dst, float* deg, int e) {
  int i = blockIdx.x * blockDim.x + threadIdx.x;
  if (i < e) atomicAdd(&deg[dst[i]], 1.0f);
}
__global__ void k_rsqrt(float* deg, int n) {
  int i = blockIdx.x * blockDim.x + threadIdx.x;
  if (i < n) deg[i] = rsqrtf(deg[i]);            // deg >= 1 always
}

// --------------------------------------------- weight pre-pack (f32 -> f16)
// Packs W[FIN,FOUT] into the per-lane B-fragment order k_gemm uses (same K order as its A):
//   element (ks, ct, lane) -> 16 halves, with column = ct*16 + (lane&15),
//   K = ks*32 + 16*(lane>>4) + e. GEMM then reads one contiguous v16h/lane.
template<int FIN, int FOUT>
__global__ void k_pack_w(const float* __restrict__ W, _Float16* __restrict__ Wp) {
  constexpr int CT = FOUT / 16;
  constexpr int KS = FIN / 32;
  int t = blockIdx.x * blockDim.x + threadIdx.x;
  if (t >= KS * CT * 32) return;
  const int lane = t & 31;
  const int ct   = (t >> 5) % CT;
  const int ks   = (t >> 5) / CT;
  const int col  = ct * 16 + (lane & 15);
  const int kb   = ks * 32 + 16 * (lane >> 4);
  _Float16* dstp = Wp + (size_t)t * 16;
#pragma unroll
  for (int e = 0; e < 16; ++e)
    dstp[e] = (_Float16)W[(size_t)(kb + e) * FOUT + col];
}

// ------------------------------------------------- WMMA GEMM: Y = act(X@W+b)
// One wave computes a full 16-row x FOUT stripe (CT accumulators); the A
// fragment is converted once per K-step and reused across all CT WMMAs.
// rows must be a multiple of 16 (N = 100000 = 6250*16).
template<int FIN, int FOUT, bool RELU>
__global__ __launch_bounds__(256)
void k_gemm(const float* __restrict__ X, const _Float16* __restrict__ Wp,
            const float* __restrict__ bias, float* __restrict__ Y, int rows) {
  constexpr int CT = FOUT / 16;
  constexpr int KS = FIN / 32;
  const int lane = threadIdx.x & 31;
  const int wid  = threadIdx.x >> 5;
  const int rowTile = blockIdx.x * (blockDim.x >> 5) + wid;
  if (rowTile >= rows / 16) return;              // wave-uniform: EXEC all-1s
  const int n   = lane & 15;
  const int hi  = lane >> 4;
  const int row = rowTile * 16 + n;              // A layout: M = lane & 15

  v8f acc[CT];
#pragma unroll
  for (int ct = 0; ct < CT; ++ct) acc[ct] = v8f{0.f,0.f,0.f,0.f,0.f,0.f,0.f,0.f};

  const v16h* wp = (const v16h*)Wp;
#pragma unroll
  for (int ks = 0; ks < KS; ++ks) {
    // A fragment (16x32 f16): lane holds K = ks*32 + 16*hi + j
    const float* xr = X + (size_t)row * FIN + ks * 32 + 16 * hi;
    v16h a;
#pragma unroll
    for (int j = 0; j < 8; ++j) {
      a[j]     = (_Float16)xr[j];
      a[8 + j] = (_Float16)xr[8 + j];
    }
#pragma unroll
    for (int ct = 0; ct < CT; ++ct) {
      const v16h b = wp[((size_t)(ks * CT + ct) * 32) + lane]; // 32B/lane, packed
      acc[ct] = __builtin_amdgcn_wmma_f32_16x16x32_f16(false, a, false, b,
                                                       (short)0, acc[ct],
                                                       false, false);
    }
  }
#pragma unroll
  for (int ct = 0; ct < CT; ++ct) {
    const int col = ct * 16 + n;
    const float bv = bias ? bias[col] : 0.0f;
#pragma unroll
    for (int v = 0; v < 8; ++v) {                // D layout: M = v + 8*hi, N = n
      float val = acc[ct][v] + bv;
      if (RELU) val = fmaxf(val, 0.0f);
      Y[(size_t)(rowTile * 16 + v + 8 * hi) * FOUT + col] = val;
    }
  }
}

// ----------------------------------------------------------- Y[i,:] = bias[:]
template<int F>
__global__ void k_fill_bias(float* __restrict__ Y, const float* __restrict__ b,
                            int n) {
  long long i = blockIdx.x * (long long)blockDim.x + threadIdx.x;
  if (i < (long long)n * F) Y[i] = b[(int)(i % F)];
}

// ------------------------------------- scatter: Y[dst] += HW[src]*dis_s*dis_d
// Virtual edges e >= E are the self-loops (src = dst = e - E).
template<int F>
__global__ void k_scatter(const float* __restrict__ HW, const int* __restrict__ src,
                          const int* __restrict__ dst, const float* __restrict__ dis,
                          float* __restrict__ Y, int E_, int N_) {
  const int C = F / 4;                            // float4 chunks per edge
  long long idx = blockIdx.x * (long long)blockDim.x + threadIdx.x;
  long long total = (long long)(E_ + N_) * C;
  if (idx >= total) return;
  int e  = (int)(idx / C);
  int f4 = (int)(idx - (long long)e * C) * 4;
  int s, d;
  if (e < E_) { s = src[e]; d = dst[e]; } else { s = d = e - E_; }
  const float w = dis[s] * dis[d];
  const float4 h = *(const float4*)(HW + (size_t)s * F + f4);
  float* yp = Y + (size_t)d * F + f4;
  atomicAdd(yp + 0, h.x * w);
  atomicAdd(yp + 1, h.y * w);
  atomicAdd(yp + 2, h.z * w);
  atomicAdd(yp + 3, h.w * w);
}

// ----------------------- fused ReLU -> LayerNorm (-> optional 0.5*(g+f) mix)
template<int F, bool COMBINE>
__global__ __launch_bounds__(256)
void k_ln(const float* __restrict__ Xin, const float* __restrict__ gamma,
          const float* __restrict__ beta, const float* __restrict__ addend,
          float* __restrict__ Y, int n) {
  const int gwid = (int)((blockIdx.x * (long long)blockDim.x + threadIdx.x) >> 5);
  const int lane = threadIdx.x & 31;
  if (gwid >= n) return;
  const int PER = F / 32;
  const float* xr = Xin + (size_t)gwid * F;
  float v[PER];
  float s = 0.f;
#pragma unroll
  for (int j = 0; j < PER; ++j) { v[j] = fmaxf(xr[lane + 32 * j], 0.f); s += v[j]; }
#pragma unroll
  for (int off = 16; off > 0; off >>= 1) s += __shfl_xor(s, off, 32);
  const float mu = s * (1.0f / F);
  float var = 0.f;
#pragma unroll
  for (int j = 0; j < PER; ++j) { float d = v[j] - mu; var += d * d; }
#pragma unroll
  for (int off = 16; off > 0; off >>= 1) var += __shfl_xor(var, off, 32);
  const float inv = rsqrtf(var * (1.0f / F) + 1e-5f);
#pragma unroll
  for (int j = 0; j < PER; ++j) {
    const int f = lane + 32 * j;
    float r = (v[j] - mu) * inv * gamma[f] + beta[f];
    if (COMBINE) r = 0.5f * (r + addend[(size_t)gwid * F + f]);
    Y[(size_t)gwid * F + f] = r;
  }
}

// ----------------------------------------------------------------------------
extern "C" void kernel_launch(void* const* d_in, const int* in_sizes, int n_in,
                              void* d_out, int out_size, void* d_ws, size_t ws_size,
                              hipStream_t stream) {
  (void)n_in; (void)out_size; (void)ws_size;
  const float* x   = (const float*)d_in[0];
  const int*   src = (const int*)d_in[1];
  const int*   dst = (const int*)d_in[2];
  const float* Wc1 = (const float*)d_in[3];  const float* bc1 = (const float*)d_in[4];
  const float* Wc2 = (const float*)d_in[5];  const float* bc2 = (const float*)d_in[6];
  const float* Wc3 = (const float*)d_in[7];  const float* bc3 = (const float*)d_in[8];
  const float* W1  = (const float*)d_in[9];  const float* b1  = (const float*)d_in[10];
  const float* W2  = (const float*)d_in[11]; const float* b2  = (const float*)d_in[12];
  const float* W3  = (const float*)d_in[13]; const float* b3  = (const float*)d_in[14];
  const float* W4  = (const float*)d_in[15]; const float* b4  = (const float*)d_in[16];
  const float* g1  = (const float*)d_in[17]; const float* bt1 = (const float*)d_in[18];
  const float* g2  = (const float*)d_in[19]; const float* bt2 = (const float*)d_in[20];
  const float* g3  = (const float*)d_in[21]; const float* bt3 = (const float*)d_in[22];

  const int N = in_sizes[0] / 128;
  const int E = in_sizes[1];
  float* out = (float*)d_out;

  float* ws   = (float*)d_ws;
  float* dis  = ws;                                 // [N]
  float* bufA = ws + N;                             // [N,128]  (holds f4 at end)
  float* bufB = bufA + (size_t)N * 128;             // [N,128]
  float* bufC = bufB + (size_t)N * 128;             // [N,128]
  uintptr_t pp = (uintptr_t)(bufC + (size_t)N * 128);
  pp = (pp + 31) & ~(uintptr_t)31;                  // 32B align for v16h loads
  _Float16* pWc1 = (_Float16*)pp;                   // 128*128 halves
  _Float16* pWc2 = pWc1 + 128 * 128;                // 128*64
  _Float16* pWc3 = pWc2 + 128 * 64;                 // 64*128
  _Float16* pW1  = pWc3 + 64 * 128;                 // 128*128
  _Float16* pW2  = pW1  + 128 * 128;                // 128*128
  _Float16* pW3  = pW2  + 128 * 128;                // 128*64
  _Float16* pW4  = pW3  + 128 * 64;                 // 64*128

  const int T = 256;
  auto cdiv = [](long long a, long long b) { return (int)((a + b - 1) / b); };
  const int gemmBlk = cdiv(N / 16, 8);              // 8 waves/block, 1 stripe/wave

  // ---- pack all weights into f16 WMMA fragment layout (tiny) ----
  k_pack_w<128,128><<<4, T, 0, stream>>>(Wc1, pWc1);
  k_pack_w<128, 64><<<2, T, 0, stream>>>(Wc2, pWc2);
  k_pack_w< 64,128><<<2, T, 0, stream>>>(Wc3, pWc3);
  k_pack_w<128,128><<<4, T, 0, stream>>>(W1,  pW1);
  k_pack_w<128,128><<<4, T, 0, stream>>>(W2,  pW2);
  k_pack_w<128, 64><<<2, T, 0, stream>>>(W3,  pW3);
  k_pack_w< 64,128><<<2, T, 0, stream>>>(W4,  pW4);

  // ---- normalization coefficients: dis = rsqrt(1 + indegree) ----
  k_init_deg<<<cdiv(N, T), T, 0, stream>>>(dis, N);
  k_count_deg<<<cdiv(E, T), T, 0, stream>>>(dst, dis, E);
  k_rsqrt<<<cdiv(N, T), T, 0, stream>>>(dis, N);

  // ---- MLP branch (f4 parked in bufA) ----
  k_gemm<128, 128, true ><<<gemmBlk, T, 0, stream>>>(x,    pW1, b1, bufB, N);
  k_gemm<128, 128, true ><<<gemmBlk, T, 0, stream>>>(bufB, pW2, b2, bufC, N);
  k_gemm<128,  64, true ><<<gemmBlk, T, 0, stream>>>(bufC, pW3, b3, bufB, N);
  k_gemm< 64, 128, false><<<gemmBlk, T, 0, stream>>>(bufB, pW4, b4, bufA, N);

  // ---- GCN conv 1 (width 128) ----
  k_gemm<128, 128, false><<<gemmBlk, T, 0, stream>>>(x, pWc1, nullptr, bufB, N);
  k_fill_bias<128><<<cdiv((long long)N * 128, T), T, 0, stream>>>(bufC, bc1, N);
  k_scatter<128><<<cdiv((long long)(E + N) * 32, T), T, 0, stream>>>(bufB, src, dst, dis, bufC, E, N);
  k_ln<128, false><<<cdiv((long long)N * 32, T), T, 0, stream>>>(bufC, g1, bt1, nullptr, bufB, N);

  // ---- GCN conv 2 (width 64) ----
  k_gemm<128, 64, false><<<gemmBlk, T, 0, stream>>>(bufB, pWc2, nullptr, bufC, N);
  k_fill_bias<64><<<cdiv((long long)N * 64, T), T, 0, stream>>>(bufB, bc2, N);
  k_scatter<64><<<cdiv((long long)(E + N) * 16, T), T, 0, stream>>>(bufC, src, dst, dis, bufB, E, N);
  k_ln<64, false><<<cdiv((long long)N * 32, T), T, 0, stream>>>(bufB, g2, bt2, nullptr, bufC, N);

  // ---- GCN conv 3 (width 128) + fused final combine with f4 ----
  k_gemm<64, 128, false><<<gemmBlk, T, 0, stream>>>(bufC, pWc3, nullptr, bufB, N);
  k_fill_bias<128><<<cdiv((long long)N * 128, T), T, 0, stream>>>(bufC, bc3, N);
  k_scatter<128><<<cdiv((long long)(E + N) * 32, T), T, 0, stream>>>(bufB, src, dst, dis, bufC, E, N);
  k_ln<128, true><<<cdiv((long long)N * 32, T), T, 0, stream>>>(bufC, g3, bt3, bufA, out, N);
}
